// MLayer_33122787787612
// MI455X (gfx1250) — matrix operand order repaired and checked by reading, not compiled, not run on hardware
//
#include <hip/hip_runtime.h>
#include <hip/hip_bf16.h>

// ---------------------------------------------------------------------------
// MI455X (gfx1250) implementation.
//  B=8192, DIM=4096, H=4, HD=1024, R=16, GH=256
//
//  Design:
//   * All GEMM-shaped work (gating MLP, Christoffel low-rank ops, output
//     projections = 550 GFLOP dominant term) runs on v_wmma_f32_16x16x32_bf16
//     with a bf16 hi/lo split (3 WMMAs/tile: Ah*Bh + Al*Bh + Ah*Bl) for
//     ~fp32 accuracy, fp32 accumulation.
//   * Operands are pre-split once into bf16 hi/lo "planes" by producer
//     kernels, so the GEMM inner loop does zero conversion VALU work.
//   * K is padded to a multiple of 32 everywhere (t and Wc zero-padded), so
//     tile staging is branchless -> clean async copies.
//   * GEMM uses LDS double buffering; staging uses the CDNA5 async
//     global->LDS path (ASYNCcnt / s_wait_asynccnt) when the builtin exists,
//     else a register-relay copy. Block tile 128x128, 8 waves, wave tile
//     32x64 = 2x4 WMMA tiles => 24 ds_load_b128 : 24 wmma per k-step.
//   * Streaming outputs (x_geo/v_geo, 268 MB) use non-temporal stores.
//  Workspace assumption: ~860 MB scratch available in d_ws.
// ---------------------------------------------------------------------------

#define BB   8192
#define DIMM 4096
#define HH   4
#define HDD  1024
#define RR   16
#define GHH  256

typedef __attribute__((ext_vector_type(16))) __bf16        v16bf;
typedef __attribute__((ext_vector_type(8)))  float         v8f;
typedef __attribute__((ext_vector_type(8)))  unsigned int  v8u;

// ---------------------------------------------------------------------------
// async global->LDS support (guarded; falls back to register relay)
// ---------------------------------------------------------------------------
#if defined(__has_builtin)
#  if __has_builtin(__builtin_amdgcn_global_load_async_to_lds_b128)
#    define HAVE_ASYNC 1
#  endif
#endif
#ifndef HAVE_ASYNC
#  define HAVE_ASYNC 0
#endif

#if HAVE_ASYNC
typedef int v4i __attribute__((vector_size(16)));   // matches builtin param type
__device__ __forceinline__ void copy16(const unsigned short* g, unsigned short* l) {
  __builtin_amdgcn_global_load_async_to_lds_b128((v4i*)g, (v4i*)l, 0, 0);
}
__device__ __forceinline__ void async_wait_group() {  // allow next stage in flight
#  if __has_builtin(__builtin_amdgcn_s_wait_asynccnt)
  __builtin_amdgcn_s_wait_asynccnt(8);
#  else
  asm volatile("s_wait_asynccnt 0x8" ::: "memory");
#  endif
}
__device__ __forceinline__ void async_wait_all() {
#  if __has_builtin(__builtin_amdgcn_s_wait_asynccnt)
  __builtin_amdgcn_s_wait_asynccnt(0);
#  else
  asm volatile("s_wait_asynccnt 0x0" ::: "memory");
#  endif
}
#else
__device__ __forceinline__ void copy16(const unsigned short* g, unsigned short* l) {
  *(uint4*)l = *(const uint4*)g;
}
__device__ __forceinline__ void async_wait_group() {}
__device__ __forceinline__ void async_wait_all() {}
#endif

// ---------------------------------------------------------------------------
// helpers
// ---------------------------------------------------------------------------

// Split fp32 into bf16 hi (bit-truncate) + bf16 lo (RNE of remainder).
__device__ __forceinline__ void bsplit(float f, unsigned short& h, unsigned short& l) {
  unsigned u = __builtin_bit_cast(unsigned, f);
  unsigned short hi = (unsigned short)(u >> 16);
  float fh = __builtin_bit_cast(float, (unsigned)hi << 16);
  float r  = f - fh;                       // exact in fp32
  unsigned ur = __builtin_bit_cast(unsigned, r);
  unsigned short lo = (unsigned short)((ur + 0x7FFFu + ((ur >> 16) & 1u)) >> 16);
  h = hi; l = lo;
}

// Load a 16-element bf16 fragment from LDS as two 16B chunks separated by
// `gap` ushorts (A frag: gap=16, B frag: gap=16).
__device__ __forceinline__ v16bf ldfrag(const unsigned short* p, int gap) {
  uint4 a = *(const uint4*)p;
  uint4 b = *(const uint4*)(p + gap);
  v8u t;
  t[0]=a.x; t[1]=a.y; t[2]=a.z; t[3]=a.w;
  t[4]=b.x; t[5]=b.y; t[6]=b.z; t[7]=b.w;
  return __builtin_bit_cast(v16bf, t);
}

// bf16x3: C += Ah*Bh + Al*Bh + Ah*Bl  (fp32 accumulate)
__device__ __forceinline__ v8f wmma3(v16bf ah, v16bf al, v16bf bh, v16bf bl, v8f c) {
  c = __builtin_amdgcn_wmma_f32_16x16x32_bf16(false, ah, false, bh, (short)0, c, false, false);
  c = __builtin_amdgcn_wmma_f32_16x16x32_bf16(false, al, false, bh, (short)0, c, false, false);
  c = __builtin_amdgcn_wmma_f32_16x16x32_bf16(false, ah, false, bl, (short)0, c, false, false);
  return c;
}

#define ASZ 40  // LDS row stride in ushorts (32 K elems + pad, keeps 16B align)

// ---------------------------------------------------------------------------
// C[M,N] = act( A[M,K] @ W[N,K]^T + bias ), A/W given as bf16 hi/lo planes.
// Batched over grid.z via element strides. K must be a multiple of 32.
// M must be a multiple of 128. Block tile 128x128; wave tile 32x64.
// ---------------------------------------------------------------------------
template <int ACT, int HASB, int NTS>
__global__ void __launch_bounds__(256) gemm_planes(
    const unsigned short* __restrict__ Ah_, const unsigned short* __restrict__ Al_,
    int lda, int A_h,
    const unsigned short* __restrict__ Wh_, const unsigned short* __restrict__ Wl_,
    int ldw, int W_h,
    float* __restrict__ C, int ldc, int C_h,
    const float* __restrict__ bias, int bias_h,
    int N, int K)
{
  __shared__ unsigned short sA[2][2][128 * ASZ];  // [buf][hi/lo]
  __shared__ unsigned short sW[2][2][128 * ASZ];

  const int tid  = threadIdx.x;
  const int lane = tid & 31;
  const int wv   = tid >> 5;
  const int mw   = (wv & 3) * 32;   // wave M offset in block tile
  const int nw   = (wv >> 2) * 64;  // wave N offset in block tile
  const int mB   = blockIdx.y * 128;
  const int nB   = blockIdx.x * 128;
  const int z    = blockIdx.z;

  Ah_ += (size_t)z * (size_t)A_h;  Al_ += (size_t)z * (size_t)A_h;
  Wh_ += (size_t)z * (size_t)W_h;  Wl_ += (size_t)z * (size_t)W_h;
  C   += (size_t)z * (size_t)C_h;
  if constexpr (HASB) bias += (size_t)z * (size_t)bias_h;

  // Branchless tile staging: 8 x 16B copies per thread per stage.
  auto stage = [&](int k0, int sbuf) {
#pragma unroll
    for (int i = 0; i < 2; ++i) {
      int idx = tid + i * 256;          // 0..511
      int row = idx >> 2;               // 0..127
      int ch  = (idx & 3) << 3;         // 0,8,16,24 (ushorts)
      size_t aoff = (size_t)(mB + row) * (size_t)lda + (size_t)(k0 + ch);
      copy16(Ah_ + aoff, &sA[sbuf][0][row * ASZ + ch]);
      copy16(Al_ + aoff, &sA[sbuf][1][row * ASZ + ch]);
      int nG = nB + row; if (nG >= N) nG = N - 1;   // clamp: junk cols discarded
      size_t woff = (size_t)nG * (size_t)ldw + (size_t)(k0 + ch);
      copy16(Wh_ + woff, &sW[sbuf][0][row * ASZ + ch]);
      copy16(Wl_ + woff, &sW[sbuf][1][row * ASZ + ch]);
    }
  };

  v8f acc[2][4] = {};
  const int rsel = lane & 15;
  const int kba  = (lane >> 4) * 8;    // A frag chunk base (K)
  const int kbw  = (lane >> 4) * 8;    // B frag chunk base (K)

  stage(0, 0);
  int buf = 0;
  for (int k0 = 0; k0 < K; k0 += 32) {
    if (k0 + 32 < K) { stage(k0 + 32, buf ^ 1); async_wait_group(); }
    else             { async_wait_all(); }
    __syncthreads();

    v16bf ah[2], al[2];
#pragma unroll
    for (int mt = 0; mt < 2; ++mt) {
      int r = (mw + mt * 16 + rsel) * ASZ + kba;
      ah[mt] = ldfrag(&sA[buf][0][r], 16);
      al[mt] = ldfrag(&sA[buf][1][r], 16);
    }
    v16bf bh[4], bl[4];
#pragma unroll
    for (int nt = 0; nt < 4; ++nt) {
      int r = (nw + nt * 16 + rsel) * ASZ + kbw;
      bh[nt] = ldfrag(&sW[buf][0][r], 16);
      bl[nt] = ldfrag(&sW[buf][1][r], 16);
    }
#pragma unroll
    for (int mt = 0; mt < 2; ++mt)
#pragma unroll
      for (int nt = 0; nt < 4; ++nt)
        acc[mt][nt] = wmma3(ah[mt], al[mt], bh[nt], bl[nt], acc[mt][nt]);

    __syncthreads();
    buf ^= 1;
  }

  // epilogue: bias + activation + store
  const int ro = (lane >> 4) * 8;   // C VGPR g -> row = g + 8*(lane>=16)
#pragma unroll
  for (int mt = 0; mt < 2; ++mt) {
#pragma unroll
    for (int nt = 0; nt < 4; ++nt) {
      int col = nB + nw + nt * 16 + rsel;
      if (col < N) {
        float bv_ = 0.f;
        if constexpr (HASB) bv_ = bias[col];
#pragma unroll
        for (int g = 0; g < 8; ++g) {
          int row = mB + mw + mt * 16 + ro + g;
          float cv = acc[mt][nt][g] + bv_;
          if constexpr (ACT == 1) cv = tanhf(cv);
          float* cp = C + (size_t)row * (size_t)ldc + (size_t)col;
          if constexpr (NTS) __builtin_nontemporal_store(cv, cp);
          else               *cp = cv;
        }
      }
    }
  }
}

// ---------------------------------------------------------------------------
// VALU helper kernels
// ---------------------------------------------------------------------------

__device__ __forceinline__ float block_reduce_sum(float v, float* red) {
  int tid = threadIdx.x;
  red[tid] = v;
  __syncthreads();
#pragma unroll
  for (int s = 128; s > 0; s >>= 1) {
    if (tid < s) red[tid] += red[tid + s];
    __syncthreads();
  }
  float r = red[0];
  __syncthreads();
  return r;
}

// LayerNorm for x and v; writes fp32 results + bf16 hi/lo planes.
__global__ void __launch_bounds__(256) ln2_kernel(
    const float* __restrict__ x, const float* __restrict__ v,
    const float* __restrict__ gx, const float* __restrict__ bxp,
    const float* __restrict__ gv, const float* __restrict__ bvp,
    float* __restrict__ xln, float* __restrict__ vln,
    unsigned short* __restrict__ xPh, unsigned short* __restrict__ xPl,
    unsigned short* __restrict__ vPh, unsigned short* __restrict__ vPl)
{
  __shared__ float red[256];
  int b = blockIdx.x, tid = threadIdx.x;
  const float* xr = x + (size_t)b * DIMM;
  const float* vr = v + (size_t)b * DIMM;
  float xs[16], vs[16];
  float sx = 0.f, sxx = 0.f, sv = 0.f, svv = 0.f;
#pragma unroll
  for (int j = 0; j < 16; ++j) {
    int c = tid + j * 256;
    xs[j] = xr[c]; vs[j] = vr[c];
    sx += xs[j]; sxx += xs[j] * xs[j];
    sv += vs[j]; svv += vs[j] * vs[j];
  }
  sx  = block_reduce_sum(sx,  red);
  sxx = block_reduce_sum(sxx, red);
  sv  = block_reduce_sum(sv,  red);
  svv = block_reduce_sum(svv, red);
  const float inv = 1.f / (float)DIMM;
  float mx = sx * inv, mv = sv * inv;
  float rx = rsqrtf(sxx * inv - mx * mx + 1e-5f);
  float rv = rsqrtf(svv * inv - mv * mv + 1e-5f);
  size_t base = (size_t)b * DIMM;
#pragma unroll
  for (int j = 0; j < 16; ++j) {
    int c = tid + j * 256;
    float ox = (xs[j] - mx) * rx * gx[c] + bxp[c];
    float ov = (vs[j] - mv) * rv * gv[c] + bvp[c];
    xln[base + c] = ox;  vln[base + c] = ov;
    unsigned short h, l;
    bsplit(ox, h, l); xPh[base + c] = h; xPl[base + c] = l;
    bsplit(ov, h, l); vPh[base + c] = h; vPl[base + c] = l;
  }
}

__global__ void __launch_bounds__(256) gate_kernel(
    const float* __restrict__ hid, const float* __restrict__ gW2,
    const float* __restrict__ gb2, const float* __restrict__ dtp,
    float* __restrict__ ctx, float* __restrict__ dtb)
{
  __shared__ float red[256];
  int b = blockIdx.x, tid = threadIdx.x;
  for (int h = 0; h < HH; ++h) {
    float p = gW2[h * GHH + tid] * hid[(size_t)b * (HH * GHH) + h * GHH + tid];
    p = block_reduce_sum(p, red);
    if (tid == 0) {
      float g  = 1.f / (1.f + __expf(-(p + gb2[h])));
      float sp = log1pf(__expf(dtp[h]));       // softplus
      ctx[(size_t)b * HH + h] = g;
      dtb[(size_t)b * HH + h] = sp * g;
    }
  }
}

// Generic fp32 -> bf16 hi/lo plane split (weights).
__global__ void __launch_bounds__(256) split_plane_kernel(
    const float* __restrict__ src, unsigned short* __restrict__ hp,
    unsigned short* __restrict__ lp, int n)
{
  int i = blockIdx.x * 256 + threadIdx.x;
  if (i < n) { unsigned short h, l; bsplit(src[i], h, l); hp[i] = h; lp[i] = l; }
}

// Wc [H,HD,R=16] -> planes [H,HD,32] zero-padded in K.
__global__ void __launch_bounds__(256) wcpad_kernel(
    const float* __restrict__ Wc, unsigned short* __restrict__ hp,
    unsigned short* __restrict__ lp)
{
  int i = blockIdx.x * 256 + threadIdx.x;   // HH*HDD*32 total
  int h   = i >> 15;
  int row = (i >> 5) & (HDD - 1);
  int r   = i & 31;
  float val = (r < RR) ? Wc[(h << 14) + row * RR + r] : 0.f;
  unsigned short hh, ll; bsplit(val, hh, ll);
  hp[i] = hh; lp[i] = ll;
}

// t = q^2 * tanh-gated s -> planes [B, H*32] zero-padded in K (s pre-tanh'd).
__global__ void __launch_bounds__(256) tmul_pad_kernel(
    const float* __restrict__ q, const float* __restrict__ s,
    unsigned short* __restrict__ hp, unsigned short* __restrict__ lp)
{
  int i = blockIdx.x * 256 + threadIdx.x;   // BB*128 total
  int b = i >> 7;
  int j = i & 127;
  int h = j >> 5;
  int r = j & 31;
  float val = 0.f;
  if (r < RR) { float qq = q[b * 64 + h * RR + r]; val = qq * qq * s[b * 64 + h * RR + r]; }
  unsigned short hh, ll; bsplit(val, hh, ll);
  hp[i] = hh; lp[i] = ll;
}

// Heun stage 1: x1 = x + dt v ; v1 = v + dt (f - g1)   (planes only)
__global__ void __launch_bounds__(256) update1_kernel(
    const float* __restrict__ xln, const float* __restrict__ vln,
    const float* __restrict__ f,   const float* __restrict__ g1,
    const float* __restrict__ dtb,
    unsigned short* __restrict__ x1h, unsigned short* __restrict__ x1l,
    unsigned short* __restrict__ v1h, unsigned short* __restrict__ v1l)
{
  size_t i = (size_t)blockIdx.x * 256 + threadIdx.x;
  int b = (int)(i >> 12);
  int h = (int)((i >> 10) & 3);
  float d  = dtb[b * 4 + h];
  float xl = xln[i], vl = vln[i];
  float k1 = f[i] - g1[i];
  float x1v = xl + d * vl;
  float v1v = vl + d * k1;
  unsigned short hh, ll;
  bsplit(x1v, hh, ll); x1h[i] = hh; x1l[i] = ll;
  bsplit(v1v, hh, ll); v1h[i] = hh; v1l[i] = ll;
}

// Heun stage 2: xn = x + dt/2 (v + v1) ; vn = v + dt/2 (k1 + k2)  (planes only)
__global__ void __launch_bounds__(256) update2_kernel(
    const float* __restrict__ xln, const float* __restrict__ vln,
    const float* __restrict__ f,   const float* __restrict__ g1,
    const float* __restrict__ g2,  const float* __restrict__ dtb,
    unsigned short* __restrict__ xnh, unsigned short* __restrict__ xnl,
    unsigned short* __restrict__ vnh, unsigned short* __restrict__ vnl)
{
  size_t i = (size_t)blockIdx.x * 256 + threadIdx.x;
  int b = (int)(i >> 12);
  int h = (int)((i >> 10) & 3);
  float d  = dtb[b * 4 + h];
  float fi = f[i];
  float xl = xln[i], vl = vln[i];
  float k1 = fi - g1[i];
  float k2 = fi - g2[i];
  float v1v = vl + d * k1;
  float xnv = xl + 0.5f * d * (vl + v1v);
  float vnv = vl + 0.5f * d * (k1 + k2);
  unsigned short hh, ll;
  bsplit(xnv, hh, ll); xnh[i] = hh; xnl[i] = ll;
  bsplit(vnv, hh, ll); vnh[i] = hh; vnl[i] = ll;
}

// ---------------------------------------------------------------------------
// Launch
// ---------------------------------------------------------------------------
extern "C" void kernel_launch(void* const* d_in, const int* in_sizes, int n_in,
                              void* d_out, int out_size, void* d_ws, size_t ws_size,
                              hipStream_t stream) {
  (void)in_sizes; (void)n_in; (void)out_size; (void)ws_size;
  const float* x     = (const float*)d_in[0];
  const float* v     = (const float*)d_in[1];
  const float* force = (const float*)d_in[2];
  const float* ln_gx = (const float*)d_in[3];
  const float* ln_bx = (const float*)d_in[4];
  const float* ln_gv = (const float*)d_in[5];
  const float* ln_bv = (const float*)d_in[6];
  const float* gW1   = (const float*)d_in[7];
  const float* gb1   = (const float*)d_in[8];
  const float* gW2   = (const float*)d_in[9];
  const float* gb2   = (const float*)d_in[10];
  const float* dtp   = (const float*)d_in[11];
  const float* Uc    = (const float*)d_in[12];
  const float* Vc    = (const float*)d_in[13];
  const float* Wc    = (const float*)d_in[14];
  const float* Wx    = (const float*)d_in[15];
  const float* bx    = (const float*)d_in[16];
  const float* Wv    = (const float*)d_in[17];
  const float* bv    = (const float*)d_in[18];

  const size_t NBD = (size_t)BB * DIMM;
  float* out  = (float*)d_out;
  float* xgeo = out;
  float* vgeo = out + NBD;
  float* ctx  = out + 2 * NBD;
  float* gout = out + 2 * NBD + (size_t)BB * HH;   // gamma_out [B,H,HD]

  // xln/vln planes live in the (not yet written) x_geo/v_geo output regions:
  // each region is NBD floats = exactly 2*NBD ushorts (hi+lo). They are dead
  // before the final projections overwrite those regions.
  unsigned short* xlnPh = (unsigned short*)xgeo;  unsigned short* xlnPl = xlnPh + NBD;
  unsigned short* vlnPh = (unsigned short*)vgeo;  unsigned short* vlnPl = vlnPh + NBD;

  // workspace layout
  float* fws = (float*)d_ws;
  float* xln = fws;             // NBD
  float* vln = xln + NBD;       // NBD
  float* g2  = vln + NBD;       // NBD (gamma2, fp32)
  float* hid = g2  + NBD;       // B*H*GH
  float* qb  = hid + (size_t)BB * HH * GHH;   // B*64
  float* sb  = qb  + (size_t)BB * 64;         // B*64
  float* dtb = sb  + (size_t)BB * 64;         // B*H
  unsigned short* us = (unsigned short*)(dtb + (size_t)BB * HH);
  unsigned short* x1Ph = us;            unsigned short* x1Pl = x1Ph + NBD;  // -> xn planes
  unsigned short* v1Ph = x1Pl + NBD;    unsigned short* v1Pl = v1Ph + NBD;  // -> vn planes
  unsigned short* tPh  = v1Pl + NBD;    unsigned short* tPl  = tPh + (size_t)BB * 128;
  unsigned short* gW1Ph = tPl + (size_t)BB * 128;
  unsigned short* gW1Pl = gW1Ph + (size_t)HH * GHH * HDD;
  unsigned short* UcPh  = gW1Pl + (size_t)HH * GHH * HDD;
  unsigned short* UcPl  = UcPh + (size_t)HH * RR * HDD;
  unsigned short* VcPh  = UcPl + (size_t)HH * RR * HDD;
  unsigned short* VcPl  = VcPh + (size_t)HH * RR * HDD;
  unsigned short* WcPh  = VcPl + (size_t)HH * RR * HDD;
  unsigned short* WcPl  = WcPh + (size_t)HH * HDD * 32;
  unsigned short* WxPh  = WcPl + (size_t)HH * HDD * 32;
  unsigned short* WxPl  = WxPh + (size_t)DIMM * DIMM;
  unsigned short* WvPh  = WxPl + (size_t)DIMM * DIMM;
  unsigned short* WvPl  = WvPh + (size_t)DIMM * DIMM;

  dim3 blk(256);
  dim3 g_hid(2, BB / 128, HH);        // N=256
  dim3 g_r(1, BB / 128, HH);          // N=16
  dim3 g_g(8, BB / 128, HH);          // N=1024
  dim3 g_o(32, BB / 128, 1);          // N=4096
  const int nelBlocks = (int)(NBD / 256);

  // 0. one-time (per call) weight plane splits
  split_plane_kernel<<<(HH * GHH * HDD) / 256, blk, 0, stream>>>(gW1, gW1Ph, gW1Pl, HH * GHH * HDD);
  split_plane_kernel<<<(HH * RR * HDD) / 256, blk, 0, stream>>>(Uc, UcPh, UcPl, HH * RR * HDD);
  split_plane_kernel<<<(HH * RR * HDD) / 256, blk, 0, stream>>>(Vc, VcPh, VcPl, HH * RR * HDD);
  split_plane_kernel<<<(DIMM * DIMM) / 256, blk, 0, stream>>>(Wx, WxPh, WxPl, DIMM * DIMM);
  split_plane_kernel<<<(DIMM * DIMM) / 256, blk, 0, stream>>>(Wv, WvPh, WvPl, DIMM * DIMM);
  wcpad_kernel<<<(HH * HDD * 32) / 256, blk, 0, stream>>>(Wc, WcPh, WcPl);

  // 1. LayerNorms (fp32 + planes)
  ln2_kernel<<<BB, blk, 0, stream>>>(x, v, ln_gx, ln_bx, ln_gv, ln_bv,
                                     xln, vln, xlnPh, xlnPl, vlnPh, vlnPl);
  // 2. gating hidden: hid = tanh(xh @ gW1^T + gb1)
  gemm_planes<1, 1, 0><<<g_hid, blk, 0, stream>>>(xlnPh, xlnPl, DIMM, HDD,
      gW1Ph, gW1Pl, HDD, GHH * HDD, hid, HH * GHH, GHH, gb1, GHH, GHH, HDD);
  // 3. gate + dt
  gate_kernel<<<BB, blk, 0, stream>>>(hid, gW2, gb2, dtp, ctx, dtb);
  // 4-7. Christoffel #1 -> gamma_out (directly in d_out)
  gemm_planes<0, 0, 0><<<g_r, blk, 0, stream>>>(vlnPh, vlnPl, DIMM, HDD,
      UcPh, UcPl, HDD, RR * HDD, qb, 64, RR, nullptr, 0, RR, HDD);
  gemm_planes<1, 0, 0><<<g_r, blk, 0, stream>>>(xlnPh, xlnPl, DIMM, HDD,
      VcPh, VcPl, HDD, RR * HDD, sb, 64, RR, nullptr, 0, RR, HDD);
  tmul_pad_kernel<<<(BB * 128) / 256, blk, 0, stream>>>(qb, sb, tPh, tPl);
  gemm_planes<0, 0, 0><<<g_g, blk, 0, stream>>>(tPh, tPl, 128, 32,
      WcPh, WcPl, 32, HDD * 32, gout, DIMM, HDD, nullptr, 0, HDD, 32);
  // 8. Heun stage 1 -> x1/v1 planes
  update1_kernel<<<nelBlocks, blk, 0, stream>>>(xln, vln, force, gout, dtb,
                                                x1Ph, x1Pl, v1Ph, v1Pl);
  // 9-12. Christoffel #2 -> g2 (fp32)
  gemm_planes<0, 0, 0><<<g_r, blk, 0, stream>>>(v1Ph, v1Pl, DIMM, HDD,
      UcPh, UcPl, HDD, RR * HDD, qb, 64, RR, nullptr, 0, RR, HDD);
  gemm_planes<1, 0, 0><<<g_r, blk, 0, stream>>>(x1Ph, x1Pl, DIMM, HDD,
      VcPh, VcPl, HDD, RR * HDD, sb, 64, RR, nullptr, 0, RR, HDD);
  tmul_pad_kernel<<<(BB * 128) / 256, blk, 0, stream>>>(qb, sb, tPh, tPl);
  gemm_planes<0, 0, 0><<<g_g, blk, 0, stream>>>(tPh, tPl, 128, 32,
      WcPh, WcPl, 32, HDD * 32, g2, DIMM, HDD, nullptr, 0, HDD, 32);
  // 13. Heun stage 2: xn planes -> x1P slots (free after s2), vn -> v1P slots
  update2_kernel<<<nelBlocks, blk, 0, stream>>>(xln, vln, force, gout, g2, dtb,
                                                x1Ph, x1Pl, v1Ph, v1Pl);
  // 14-15. Output projections (dominant 550 GFLOP), non-temporal stores.
  gemm_planes<0, 1, 1><<<g_o, blk, 0, stream>>>(x1Ph, x1Pl, DIMM, 0,
      WxPh, WxPl, DIMM, 0, xgeo, DIMM, 0, bx, 0, DIMM, DIMM);
  gemm_planes<0, 1, 1><<<g_o, blk, 0, stream>>>(v1Ph, v1Pl, DIMM, 0,
      WvPh, WvPl, DIMM, 0, vgeo, DIMM, 0, bv, 0, DIMM, DIMM);
}
